// GCN_L_21827023798522
// MI455X (gfx1250) — matrix operand order repaired and checked by reading, not compiled, not run on hardware
//
#include <hip/hip_runtime.h>

// ---------------------------------------------------------------------------
// CDNA5 / gfx1250 GCN pipeline.  Dominant cost: 8192^3 GEMM -> bf16 WMMA.
// bf16-A GEMMs stage the A tile with the Tensor Data Mover (async, TENSORcnt).
// ---------------------------------------------------------------------------

typedef __attribute__((ext_vector_type(16))) __bf16 v16bf;
typedef __attribute__((ext_vector_type(8)))  __bf16 v8bf;
typedef __attribute__((ext_vector_type(2)))  __bf16 v2bf;
typedef __attribute__((ext_vector_type(8)))  float  v8f;
typedef __attribute__((ext_vector_type(4)))  float  v4f;
typedef __attribute__((ext_vector_type(4)))  unsigned int u32x4;
typedef __attribute__((ext_vector_type(8)))  int i32x8;
typedef __attribute__((ext_vector_type(4)))  int i32x4;

#define BM 128
#define BN 128
#define BK 32
#define GROUP_M 8   // L2 supertile: 8 consecutive blocks share one B panel

// Issue one TDM descriptor: copy a BM x BK bf16 tile (row-major, stride
// `stride_elems`) starting at `gptr` into LDS at byte offset `lds_addr`.
// D# layout per CDNA5 ISA sec 8.3/8.4 (2-D tile, groups 2/3 zero).
__device__ __forceinline__ void tdm_load_tile_bf16(unsigned lds_addr, const void* gptr,
                                                   unsigned td0, unsigned td1,
                                                   unsigned stride_elems)
{
    unsigned long long ga = (unsigned long long)(size_t)gptr;
    u32x4 g0;
    g0[0] = 1u;                                        // count=1, user descriptor
    g0[1] = lds_addr;                                  // LDS byte address
    g0[2] = (unsigned)ga;                              // global_addr[31:0]
    g0[3] = (unsigned)((ga >> 32) & 0x01FFFFFFu) | (2u << 30);  // addr[56:32], type=2
    i32x8 g1;
    g1[0] = 1 << 16;                                   // data_size = 2 bytes
    g1[1] = (int)((td0 & 0xFFFFu) << 16);              // tensor_dim0[15:0]
    g1[2] = (int)((td0 >> 16) | ((td1 & 0xFFFFu) << 16));   // dim0 hi | dim1 lo
    g1[3] = (int)((td1 >> 16) | ((unsigned)BK << 16)); // dim1 hi | tile_dim0 = 32
    g1[4] = BM;                                        // tile_dim1 = 128, tile_dim2 = 0
    g1[5] = (int)stride_elems;                         // tensor_dim0_stride[31:0]
    g1[6] = 0;                                         // stride hi | dim1_stride lo
    g1[7] = 0;
    i32x4 z4; z4[0] = 0; z4[1] = 0; z4[2] = 0; z4[3] = 0;
    i32x8 z8;
    #pragma unroll
    for (int i = 0; i < 8; i++) z8[i] = 0;
    // 6-arg form (clang-23 / therock headers): (g0, g1, g2, g3, g4, cpol)
    __builtin_amdgcn_tensor_load_to_lds(g0, g1, z4, z4, z8, 0);
}

// ATYPE: 0 = A is f32 (convert to bf16 at staging), 1 = A is bf16 (TDM-staged).
// RELU/BIAS: fused epilogue.  OBF16: store C as bf16 (else f32).
template<int ATYPE, bool RELU, bool BIAS, bool OBF16>
__global__ __launch_bounds__(256) void gemm_wmma_k(
    const void* __restrict__ Av, const float* __restrict__ B,
    const float* __restrict__ bias, void* __restrict__ Cv,
    int M, int N, int K, int lda, int ldb, int ldc)
{
    constexpr int NBUF = (ATYPE == 1) ? 2 : 1;         // double-buffer A for TDM
    __shared__ alignas(16) __bf16 As[NBUF][BM][BK];    // [m][k]
    __shared__ alignas(16) __bf16 Bt[BN][BK];          // [n][k] (transposed)

    // ---- L2-friendly block swizzle (triton GROUP_M scheme) ----
    const int grid_m = gridDim.y, grid_n = gridDim.x;
    int pid = blockIdx.y * grid_n + blockIdx.x;
    int width   = GROUP_M * grid_n;
    int group   = pid / width;
    int first_m = group * GROUP_M;
    int size_m  = (grid_m - first_m) < GROUP_M ? (grid_m - first_m) : GROUP_M;
    int pid_m   = first_m + (pid % size_m);
    int pid_n   = (pid % width) / size_m;

    const int tid  = threadIdx.x;
    const int m0   = pid_m * BM;
    const int n0b  = pid_n * BN;

    const int lane = tid & 31;
    const int wave = tid >> 5;
    const int g    = lane >> 4;          // half-wave group (K-split in fragments)
    const int ln   = lane & 15;          // row/col within a 16-wide tile
    const int wm   = (wave >> 1) * 32;   // wave M offset inside block tile
    const int wn   = (wave & 1) * 64;    // wave N offset inside block tile

    v8f acc[2][4];
    #pragma unroll
    for (int a = 0; a < 2; a++)
        #pragma unroll
        for (int b = 0; b < 4; b++)
            #pragma unroll
            for (int r = 0; r < 8; r++) acc[a][b][r] = 0.0f;

    // B staging mapping: thread -> (pair of k rows, 8-wide n segment)
    const int kp = tid >> 4;             // 0..15 -> k rows {2kp, 2kp+1}
    const int sn = (tid & 15) * 8;       // 0..120

    // Prologue: kick off the first A tile DMA (wave 0 only; EXEC-independent).
    if constexpr (ATYPE == 1) {
        if (wave == 0) {
            tdm_load_tile_bf16((unsigned)(size_t)&As[0][0][0],
                               (const __bf16*)Av + (size_t)m0 * lda,
                               (unsigned)K, (unsigned)M, (unsigned)lda);
        }
    }

    int ib = 0;
    for (int k0 = 0; k0 < K; k0 += BK, ib ^= 1) {
        // ---- stage A tile (f32 path only): BM x BK -> bf16 LDS ----
        if constexpr (ATYPE == 0) {
            #pragma unroll
            for (int c = 0; c < 2; c++) {
                int idx = tid + c * 256;           // 512 chunks of 8 elements
                int am  = idx >> 2;
                int ak  = (idx & 3) * 8;
                __bf16* dst = &As[0][am][ak];
                const float* src = (const float*)Av + (size_t)(m0 + am) * lda + k0 + ak;
                v4f f0 = *(const v4f*)src;
                v4f f1 = *(const v4f*)(src + 4);
                #pragma unroll
                for (int i = 0; i < 4; i++) dst[i]     = (__bf16)f0[i];
                #pragma unroll
                for (int i = 0; i < 4; i++) dst[4 + i] = (__bf16)f1[i];
            }
        }
        // ---- stage B tile (BK x BN) transposed -> Bt[n][k], packed b32 ----
        {
            const float* r0 = B + (size_t)(k0 + 2 * kp) * ldb;
            const float* r1 = r0 + ldb;
            float a0[8], a1[8];
            #pragma unroll
            for (int i = 0; i < 8; i += 4) {
                int col = n0b + sn + i;
                v4f f0, f1;
                if (col < N) { f0 = *(const v4f*)(r0 + col); f1 = *(const v4f*)(r1 + col); }
                else {
                    f0[0]=f0[1]=f0[2]=f0[3]=0.f;
                    f1[0]=f1[1]=f1[2]=f1[3]=0.f;
                }
                #pragma unroll
                for (int t = 0; t < 4; t++) { a0[i+t] = f0[t]; a1[i+t] = f1[t]; }
            }
            #pragma unroll
            for (int i = 0; i < 8; i++) {
                v2bf pr; pr[0] = (__bf16)a0[i]; pr[1] = (__bf16)a1[i];
                *(v2bf*)&Bt[sn + i][2 * kp] = pr;    // one ds_store_b32
            }
        }
        // ---- TDM pipeline control (bf16-A path) ----
        if constexpr (ATYPE == 1) {
            if (wave == 0) {
                if (k0 + BK < K) {
                    tdm_load_tile_bf16((unsigned)(size_t)&As[ib ^ 1][0][0],
                                       (const __bf16*)Av + (size_t)m0 * lda + (k0 + BK),
                                       (unsigned)K, (unsigned)M, (unsigned)lda);
                    __builtin_amdgcn_s_wait_tensorcnt(1);  // current tile done
                } else {
                    __builtin_amdgcn_s_wait_tensorcnt(0);
                }
            }
        }
        __syncthreads();

        // ---- load fragments (ISA 16-bit A 16x32 / B 32x16 layouts) ----
        union { v16bf v; v8bf h[2]; } af[2], bfrag[4];
        #pragma unroll
        for (int tm = 0; tm < 2; tm++) {
            // lane row = M; halves 0-7 = K g*8..g*8+7, halves 8-15 = K 16+g*8..
            const __bf16* ap = &As[ib][wm + tm * 16 + ln][g * 8];
            af[tm].h[0] = *(const v8bf*)ap;
            af[tm].h[1] = *(const v8bf*)(ap + 16);
        }
        #pragma unroll
        for (int tn = 0; tn < 4; tn++) {
            // lane col = N; halves 0-7 = K g*8..g*8+7, halves 8-15 = K 16+g*8..
            const __bf16* bp = &Bt[wn + tn * 16 + ln][g * 8];
            bfrag[tn].h[0] = *(const v8bf*)bp;
            bfrag[tn].h[1] = *(const v8bf*)(bp + 16);
        }
        #pragma unroll
        for (int tm = 0; tm < 2; tm++)
            #pragma unroll
            for (int tn = 0; tn < 4; tn++)
                acc[tm][tn] = __builtin_amdgcn_wmma_f32_16x16x32_bf16(
                    false, af[tm].v, false, bfrag[tn].v,
                    (short)0, acc[tm][tn], false, false);
        __syncthreads();
    }

    // ---- epilogue: bias + relu + store ----
    float bcol[4];
    #pragma unroll
    for (int tn = 0; tn < 4; tn++) {
        int col = n0b + wn + tn * 16 + ln;
        bcol[tn] = (BIAS && col < N) ? bias[col] : 0.0f;
    }
    #pragma unroll
    for (int tm = 0; tm < 2; tm++) {
        #pragma unroll
        for (int tn = 0; tn < 4; tn++) {
            int col = n0b + wn + tn * 16 + ln;
            if (col < N) {
                #pragma unroll
                for (int r = 0; r < 8; r++) {
                    float v = acc[tm][tn][r] + bcol[tn];
                    if (RELU) v = fmaxf(v, 0.0f);
                    size_t off = (size_t)(m0 + wm + tm * 16 + g * 8 + r) * ldc + col;
                    if (OBF16) ((__bf16*)Cv)[off] = (__bf16)v;
                    else       ((float*)Cv)[off]  = v;
                }
            }
        }
    }
}

// Deterministic per-row sum of the 8192x8192 bf16 matrix (block per row).
__global__ __launch_bounds__(256) void k_rowsum(const void* __restrict__ av,
                                                float* __restrict__ rowsum)
{
    const __bf16* a = (const __bf16*)av + ((size_t)blockIdx.x << 13);
    float s = 0.0f;
    for (int c = threadIdx.x; c < 1024; c += 256) {       // 1024 chunks of 8
        v8bf v = *(const v8bf*)(a + c * 8);
        #pragma unroll
        for (int t = 0; t < 8; t++) s += (float)v[t];
    }
    __shared__ float red[256];
    red[threadIdx.x] = s;
    __syncthreads();
    for (int o = 128; o > 0; o >>= 1) {
        if (threadIdx.x < o) red[threadIdx.x] += red[threadIdx.x + o];
        __syncthreads();
    }
    if (threadIdx.x == 0) rowsum[blockIdx.x] = red[0];
}

__global__ void k_dinv(const float* __restrict__ rowsum, float* __restrict__ dinv, int n)
{
    int i = blockIdx.x * blockDim.x + threadIdx.x;
    if (i < n) dinv[i] = rsqrtf(rowsum[i] + 1.0f);   // +1 for the identity diagonal
}

// In-place: a[i][j] = dinv[i] * (relu_val + (i==j)) * dinv[j], bf16, N=8192.
__global__ __launch_bounds__(256) void k_norm(void* __restrict__ av,
                                              const float* __restrict__ dinv)
{
    size_t chunk = (size_t)blockIdx.x * 256 + threadIdx.x;   // 8192*1024 chunks
    int i  = (int)(chunk >> 10);
    int j0 = (int)(chunk & 1023) << 3;
    float di = dinv[i];
    __bf16* a = (__bf16*)av;
    v8bf* p = (v8bf*)(a + ((size_t)i << 13) + j0);
    v8bf v = *p;
    #pragma unroll
    for (int t = 0; t < 8; t++) {
        int j = j0 + t;
        float f = (float)v[t] + (j == i ? 1.0f : 0.0f);
        v[t] = (__bf16)(f * di * dinv[j]);
    }
    *p = v;
}

// In-place log-softmax over rows of [M, N] (N = 40, thread per row).
__global__ void k_logsoftmax(float* __restrict__ out, int M, int N)
{
    int r = blockIdx.x * blockDim.x + threadIdx.x;
    if (r >= M) return;
    float* p = out + (size_t)r * N;
    float mx = -3.4e38f;
    for (int j = 0; j < N; j++) mx = fmaxf(mx, p[j]);
    float s = 0.0f;
    for (int j = 0; j < N; j++) s += expf(p[j] - mx);
    float ls = logf(s);
    for (int j = 0; j < N; j++) p[j] = p[j] - mx - ls;
}

extern "C" void kernel_launch(void* const* d_in, const int* in_sizes, int n_in,
                              void* d_out, int out_size, void* d_ws, size_t ws_size,
                              hipStream_t stream)
{
    const int N = 8192, F = 512, H = 256, C = 40;
    const float* x   = (const float*)d_in[0];
    const float* adj = (const float*)d_in[1];
    const float* sw  = (const float*)d_in[2];
    const float* sb  = (const float*)d_in[3];
    const float* W0  = (const float*)d_in[4];
    const float* b0  = (const float*)d_in[5];
    const float* W1  = (const float*)d_in[6];
    const float* b1  = (const float*)d_in[7];
    const float* W2  = (const float*)d_in[8];
    const float* b2  = (const float*)d_in[9];
    float* out = (float*)d_out;

    // workspace layout (~145 MB)
    char* w = (char*)d_ws;
    void*  modbuf = (void*)w;                                   // N*N bf16 = 128 MB
    float* rowsum = (float*)(w + (size_t)N * N * 2);            // 32 KB
    float* dinv   = rowsum + N;                                 // 32 KB
    float* H0     = dinv + N;                                   // N*H f32 = 8 MB
    float* H1     = H0 + (size_t)N * H;                         // 8 MB

    dim3 blk(256);

    // 1) mod = relu(adj @ struct_weight + sb)  -> bf16   [1.1 TFLOP, compute-bound]
    gemm_wmma_k<0, true, true, true><<<dim3(64, 64), blk, 0, stream>>>(
        adj, sw, sb, modbuf, N, N, N, N, N, N);

    // 2) symmetric normalization of (mod + I)
    k_rowsum<<<N, 256, 0, stream>>>(modbuf, rowsum);
    k_dinv<<<32, 256, 0, stream>>>(rowsum, dinv, N);
    k_norm<<<32768, 256, 0, stream>>>(modbuf, dinv);

    // 3) GCN layer 0: H0 = x @ W0 ; H1 = relu(adjn @ H0 + b0)
    gemm_wmma_k<0, false, false, false><<<dim3(2, 64), blk, 0, stream>>>(
        x, W0, nullptr, H0, N, H, F, F, H, H);
    gemm_wmma_k<1, true, true, false><<<dim3(2, 64), blk, 0, stream>>>(
        modbuf, H0, b0, H1, N, H, N, N, H, H);

    // 4) GCN layer 1
    gemm_wmma_k<0, false, false, false><<<dim3(2, 64), blk, 0, stream>>>(
        H1, W1, nullptr, H0, N, H, H, H, H, H);
    gemm_wmma_k<1, true, true, false><<<dim3(2, 64), blk, 0, stream>>>(
        modbuf, H0, b1, H1, N, H, N, N, H, H);

    // 5) GCN layer 2 (no relu) -> d_out
    gemm_wmma_k<0, false, false, false><<<dim3(1, 64), blk, 0, stream>>>(
        H1, W2, nullptr, H0, N, C, H, H, C, C);
    gemm_wmma_k<1, false, true, false><<<dim3(1, 64), blk, 0, stream>>>(
        modbuf, H0, b2, out, N, C, N, N, C, C);

    // 6) log_softmax rows
    k_logsoftmax<<<32, 256, 0, stream>>>(out, N, C);
}
